// Mex_2534030705206
// MI455X (gfx1250) — matrix operand order repaired and checked by reading, not compiled, not run on hardware
//
#include <hip/hip_runtime.h>
#include <hip/hip_bf16.h>

typedef __attribute__((ext_vector_type(16))) _Float16 v16h;
typedef __attribute__((ext_vector_type(8)))  _Float16 v8h;
typedef __attribute__((ext_vector_type(8)))  float    v8f;
typedef __attribute__((ext_vector_type(4)))  float    v4f;

#define MEX_B   64
#define MEX_C   64
#define MEX_H   64
#define MEX_W   64
#define MEX_NI  256
#define MEX_K   576          // 64*3*3
#define MEX_OH  32
#define MEX_OW  32
#define EPS     1.0f
#define LOGK    6.3561077f   // ln(576)

#define M_TILE   128         // patch rows per workgroup (8 waves x 16)
#define A_STRIDE 584         // halfs/row: 576+8 pad -> 1168B (16B aligned, 36 dw step => 16 banks)
#define B_STRIDE 40          // halfs/row: 32+8 pad  -> 80B  (16B aligned, 20 dw step => 16 banks)
#define NKB      18          // 576/32 K-steps

#define A_BYTES   (M_TILE * A_STRIDE * 2)
#define B_BYTES   (MEX_NI * B_STRIDE * 2)
#define MP_OFF    (A_BYTES + B_BYTES)
#define PART_OFF  (MP_OFF + M_TILE * 4)
#define SMEM_BYTES (PART_OFF + 256 * 4)   // ~171.5 KB dynamic LDS (<= 320 KB/WGP)

// ---------------------------------------------------------------------------
// Prep: per instance i compute mo[i] = eps*max_k O[i][k] and
// Oexp[i][k] = f16( exp(eps*O[i][k] - mo[i]) ).  One wave32 per instance.
// ---------------------------------------------------------------------------
__global__ __launch_bounds__(32)
void mex_prep(const float* __restrict__ offsets, _Float16* __restrict__ Oexp,
              float* __restrict__ mo)
{
    const int n    = blockIdx.x;
    const int lane = threadIdx.x;
    const float* row = offsets + n * MEX_K;

    float mx = -3.4e38f;
    for (int k = lane; k < MEX_K; k += 32) mx = fmaxf(mx, row[k]);
    #pragma unroll
    for (int off = 16; off > 0; off >>= 1) mx = fmaxf(mx, __shfl_xor(mx, off, 32));
    const float me = EPS * mx;
    if (lane == 0) mo[n] = me;

    for (int k = lane; k < MEX_K; k += 32)
        Oexp[n * MEX_K + k] = (_Float16)__expf(EPS * row[k] - me);
}

// ---------------------------------------------------------------------------
// Main: fused im2col + exp + f16 WMMA GEMM + log epilogue.
//   M = 65536 patches, N = 256 instances, K = 576.
// ---------------------------------------------------------------------------
__global__ __launch_bounds__(256, 1)
void mex_main(const float* __restrict__ x, const _Float16* __restrict__ Oexp,
              const float* __restrict__ mo, float* __restrict__ out)
{
    extern __shared__ char smem[];
    _Float16* Alds = (_Float16*)smem;                 // [128][584] f16
    _Float16* Blds = (_Float16*)(smem + A_BYTES);     // [256][40]  f16 (K-slice of Oexp)
    float*    mpL  = (float*)(smem + MP_OFF);         // [128] eps*rowmax
    float*    part = (float*)(smem + PART_OFF);       // [256] partial maxes, then mo[] cache

    const int t    = threadIdx.x;
    const int lane = t & 31;
    const int wv   = t >> 5;
    const int m0   = blockIdx.x * M_TILE;

    // thread -> (row r, K-half kh): 2 threads per row, 32 channels each
    const int r  = t >> 1;
    const int kh = t & 1;
    const int m  = m0 + r;
    const int pb  = m >> 10;          // batch
    const int oh  = (m >> 5) & 31;
    const int ow  = m & 31;
    const int ih0 = oh * 2 - 1;       // pad 1, stride 2
    const int iw0 = ow * 2 - 1;
    const int c0  = kh * 32;

    // ---- Phase 1: row max over 576-entry window (OOB contributes 0.0, matching
    //      zero-padded patches); stash eps*x as f16 into the A tile while at it
    //      so x is read from global exactly once. ----
    {
        float mx = -3.4e38f;
        _Float16* arow = Alds + r * A_STRIDE;
        for (int c = c0; c < c0 + 32; ++c) {
            const float* xc = x + ((pb * MEX_C + c) * MEX_H) * MEX_W;
            const int kb = c * 9;
            #pragma unroll
            for (int fh = 0; fh < 3; ++fh) {
                const int ih = ih0 + fh;
                const bool okh = (unsigned)ih < (unsigned)MEX_H;
                #pragma unroll
                for (int fw = 0; fw < 3; ++fw) {
                    const int iw = iw0 + fw;
                    const float v = (okh && (unsigned)iw < (unsigned)MEX_W)
                                        ? xc[ih * MEX_W + iw] : 0.0f;
                    mx = fmaxf(mx, v);
                    arow[kb + fh * 3 + fw] = (_Float16)(EPS * v);
                }
            }
        }
        part[t] = mx;
    }
    __syncthreads();
    if (t < M_TILE) mpL[t] = EPS * fmaxf(part[2 * t], part[2 * t + 1]);
    __syncthreads();
    part[t] = mo[t];        // repurpose part[] as an LDS cache of mo[256]

    // ---- Phase 2: in-place A = f16(exp(a - mp)), vectorized v8h sweep ----
    {
        const float mp = mpL[r];
        v8h* ar8 = (v8h*)(Alds + r * A_STRIDE + kh * 288);   // 288 halfs, 16B aligned
        for (int i = 0; i < 36; ++i) {
            v8h a = ar8[i];
            #pragma unroll
            for (int j = 0; j < 8; ++j)
                a[j] = (_Float16)__expf((float)a[j] - mp);
            ar8[i] = a;
        }
    }

    // ---- Phase 3: GEMM. acc[nt] covers N-tile nt (16 cols), this wave's 16 rows.
    // Fragment layouts per ISA 7.12.2 (wave32):
    //   A 16x32 f16: lane<16 -> M=lane, K={0..7,16..23}; lane>=16 -> M=lane-16, K={8..15,24..31}
    //   B 32x16 f16: lane<16 -> N=lane, K={0..7,16..23};  lane>=16 -> N=lane-16, K={8..15,24..31}
    v8f acc[16];
    {
        v8f z = {};
        #pragma unroll
        for (int nt = 0; nt < 16; ++nt) acc[nt] = z;
    }

    union Frag { v16h v; v8h h[2]; };
    const int arow_i = wv * 16 + (lane & 15);
    const int akoff  = (lane < 16) ? 0 : 8;
    const int bn16   = lane & 15;
    const int bkoff  = (lane < 16) ? 0 : 8;

    auto loadB = [&](int nt, Frag& f) {
        const _Float16* bp = Blds + (nt * 16 + bn16) * B_STRIDE + bkoff;
        f.h[0] = *(const v8h*)(bp);
        f.h[1] = *(const v8h*)(bp + 16);
    };

    // register double-buffer for the global B K-slice (overlaps fetch w/ compute)
    v8h stg[2][4];
    {
        const v8h* src = (const v8h*)(Oexp + t * MEX_K);
        stg[0][0] = src[0]; stg[0][1] = src[1]; stg[0][2] = src[2]; stg[0][3] = src[3];
    }

    for (int kb = 0; kb < NKB; ++kb) {
        const int cur = kb & 1;
        __syncthreads();   // previous-iteration B reads done (and phases 1-2 on iter 0)
        {
            _Float16* dst = Blds + t * B_STRIDE;
            *(v8h*)(dst + 0)  = stg[cur][0];  *(v8h*)(dst + 8)  = stg[cur][1];
            *(v8h*)(dst + 16) = stg[cur][2];  *(v8h*)(dst + 24) = stg[cur][3];
        }
        if (kb + 1 < NKB) {   // prefetch next K-slice into the other register set
            const v8h* src = (const v8h*)(Oexp + t * MEX_K + (kb + 1) * 32);
            stg[cur ^ 1][0] = src[0]; stg[cur ^ 1][1] = src[1];
            stg[cur ^ 1][2] = src[2]; stg[cur ^ 1][3] = src[3];
        }
        __syncthreads();

        Frag af;
        const _Float16* ap = Alds + arow_i * A_STRIDE + kb * 32 + akoff;
        af.h[0] = *(const v8h*)(ap);        // K 0..7  (or 8..15)
        af.h[1] = *(const v8h*)(ap + 16);   // K 16..23 (or 24..31)

        Frag bf[2];
        loadB(0, bf[0]);
        #pragma unroll
        for (int nt = 0; nt < 16; ++nt) {
            if (nt + 1 < 16) loadB(nt + 1, bf[(nt + 1) & 1]);   // ping-pong B frags
            acc[nt] = __builtin_amdgcn_wmma_f32_16x16x32_f16(
                false, af.v, false, bf[nt & 1].v, (short)0, acc[nt], false, false);
        }
    }

    // ---- Phase 4: epilogue  y = (log S + mp + mo)/eps - ln(K)/eps ----
    // C layout: VGPR g -> M = g (lanes<16) / g+8 (lanes>=16); N = lane%16.
    // Consecutive g => consecutive ow (base 8-aligned, no 32-wrap) => b128 stores.
    const int rbase = wv * 16 + ((lane < 16) ? 0 : 8);
    float mpv[8];
    #pragma unroll
    for (int g = 0; g < 8; ++g) mpv[g] = mpL[rbase + g];

    const int mmb  = m0 + rbase;            // multiple of 8
    const int ob   = mmb >> 10;
    const int ooh  = (mmb >> 5) & 31;
    const int oowb = mmb & 31;
    float* obase = out + (ob << 18) + (ooh << 5) + oowb;

    #pragma unroll
    for (int nt = 0; nt < 16; ++nt) {
        const int n = nt * 16 + bn16;
        const float mon = part[n];          // LDS-cached mo
        float* orow = obase + (n << 10);
        v4f lo, hi;
        #pragma unroll
        for (int g = 0; g < 4; ++g) {
            lo[g] = (__logf(acc[nt][g])     + mpv[g]     + mon) * (1.0f / EPS) - LOGK / EPS;
            hi[g] = (__logf(acc[nt][g + 4]) + mpv[g + 4] + mon) * (1.0f / EPS) - LOGK / EPS;
        }
        *(v4f*)(orow)     = lo;
        *(v4f*)(orow + 4) = hi;
    }
}

// ---------------------------------------------------------------------------
extern "C" void kernel_launch(void* const* d_in, const int* in_sizes, int n_in,
                              void* d_out, int out_size, void* d_ws, size_t ws_size,
                              hipStream_t stream) {
    const float* x       = (const float*)d_in[0];   // (64,64,64,64) f32
    const float* offsets = (const float*)d_in[1];   // (1,256,64,3,3) f32
    float*       out     = (float*)d_out;           // (64,256,32,32) f32

    // workspace: Oexp f16 [256][576] then mo f32 [256]  (~290 KB)
    _Float16* Oexp = (_Float16*)d_ws;
    float*    mo   = (float*)((char*)d_ws + (size_t)MEX_NI * MEX_K * sizeof(_Float16));

    mex_prep<<<MEX_NI, 32, 0, stream>>>(offsets, Oexp, mo);

    const int ngrp = (MEX_B * MEX_OH * MEX_OW) / M_TILE;   // 512
    mex_main<<<ngrp, 256, SMEM_BYTES, stream>>>(x, Oexp, mo, out);
}
